// MultiHeadAttention_83966610637570
// MI455X (gfx1250) — matrix operand order repaired and checked by reading, not compiled, not run on hardware
//
#include <hip/hip_runtime.h>
#include <cstdint>

// Problem constants (from reference)
#define B_   2
#define S_   2048
#define D_   1024
#define H_   16
#define DH_  64
#define SCALE 0.125f   // 1/sqrt(64)

typedef __attribute__((ext_vector_type(16))) _Float16 v16h;
typedef __attribute__((ext_vector_type(8)))  _Float16 v8h;
typedef __attribute__((ext_vector_type(8)))  float    v8f;

static __device__ __forceinline__ v16h mk16(v8h lo, v8h hi) {
  return __builtin_shufflevector(lo, hi, 0,1,2,3,4,5,6,7,8,9,10,11,12,13,14,15);
}
static __device__ __forceinline__ v8f wmma_f16(v16h a, v16h b, v8f c) {
  // D = A(16x32 f16) * B(32x16 f16) + C(16x16 f32)
  return __builtin_amdgcn_wmma_f32_16x16x32_f16(false, a, false, b, (short)0, c,
                                                false, false);
}
static __device__ __forceinline__ v8f zero8() {
  v8f z = {0.f, 0.f, 0.f, 0.f, 0.f, 0.f, 0.f, 0.f};
  return z;
}

// ---------------------------------------------------------------------------
// 1) LayerNorm (fp32 in) -> xn (f16), one block per row of 1024
// ---------------------------------------------------------------------------
__global__ __launch_bounds__(256) void ln_kernel(const float* __restrict__ x,
                                                 const float* __restrict__ g,
                                                 const float* __restrict__ bta,
                                                 _Float16* __restrict__ xnh) {
  __shared__ float sred[2][8];
  const int row = blockIdx.x;
  const int t = threadIdx.x;
  const float* xr = x + (size_t)row * D_;
  float xv[4];
  float s = 0.f, s2 = 0.f;
#pragma unroll
  for (int i = 0; i < 4; ++i) {
    xv[i] = xr[t + i * 256];
    s += xv[i];
    s2 += xv[i] * xv[i];
  }
#pragma unroll
  for (int d = 16; d >= 1; d >>= 1) {
    s  += __shfl_xor(s,  d, 32);
    s2 += __shfl_xor(s2, d, 32);
  }
  if ((t & 31) == 0) { sred[0][t >> 5] = s; sred[1][t >> 5] = s2; }
  __syncthreads();
  float ts = 0.f, ts2 = 0.f;
#pragma unroll
  for (int w = 0; w < 8; ++w) { ts += sred[0][w]; ts2 += sred[1][w]; }
  const float mu   = ts * (1.0f / D_);
  const float var  = ts2 * (1.0f / D_) - mu * mu;
  const float rstd = rsqrtf(var + 1e-5f);
#pragma unroll
  for (int i = 0; i < 4; ++i) {
    const int c = t + i * 256;
    xnh[(size_t)row * D_ + c] = (_Float16)((xv[i] - mu) * rstd * g[c] + bta[c]);
  }
}

// ---------------------------------------------------------------------------
// 2) fp32 -> f16 weight conversion
// ---------------------------------------------------------------------------
__global__ void cvt_kernel(const float* __restrict__ w, _Float16* __restrict__ wh,
                           int n) {
  int i = blockIdx.x * blockDim.x + threadIdx.x;
  const int stride = gridDim.x * blockDim.x;
  for (; i < n; i += stride) wh[i] = (_Float16)w[i];
}

// ---------------------------------------------------------------------------
// 3) QKV GEMM: y = xn @ W^T + b  (per wave: 16x64 strip, K=1024)
//    blockIdx.y selects Q/K/V. Q,K stored [b,h,s,64]; V stored transposed
//    [b,h,64,s] so attention's P*V B-fragments are contiguous loads.
// ---------------------------------------------------------------------------
__global__ __launch_bounds__(256) void qkv_gemm_kernel(
    const _Float16* __restrict__ xnh,
    const _Float16* __restrict__ wqh, const _Float16* __restrict__ wkh,
    const _Float16* __restrict__ wvh,
    const float* __restrict__ bq, const float* __restrict__ bk,
    const float* __restrict__ bv,
    _Float16* __restrict__ qh, _Float16* __restrict__ kh,
    _Float16* __restrict__ vth) {
  const int which = blockIdx.y;
  const _Float16* W = (which == 0) ? wqh : (which == 1) ? wkh : wvh;
  const float* bias = (which == 0) ? bq : (which == 1) ? bk : bv;

  const int wid = threadIdx.x >> 5, lane = threadIdx.x & 31;
  const int lq = lane & 15, hi = lane >> 4;
  const int job = blockIdx.x * 8 + wid;      // 0..4095
  const int mt = job >> 4, ns = job & 15;    // 256 M-tiles x 16 N-strips
  const int m  = mt * 16 + lq;               // A-fragment row
  const int n0 = ns * 64;

  v8f acc[4] = {zero8(), zero8(), zero8(), zero8()};
  const _Float16* arow = xnh + (size_t)m * D_;

  for (int kt = 0; kt < D_; kt += 32) {
    const _Float16* ap = arow + kt + hi * 8;
    __builtin_prefetch(ap + 128, 0, 0);              // global_prefetch_b8
    v16h a = mk16(*(const v8h*)ap, *(const v8h*)(ap + 16));
#pragma unroll
    for (int j = 0; j < 4; ++j) {
      const int n = n0 + j * 16 + lq;                // B-fragment column
      v16h b = mk16(*(const v8h*)(W + (size_t)n * D_ + kt + hi * 8), *(const v8h*)(W + (size_t)n * D_ + kt + hi * 8 + 16));
      acc[j] = wmma_f16(a, b, acc[j]);
    }
  }

#pragma unroll
  for (int j = 0; j < 4; ++j) {
    const int n = n0 + j * 16 + lq;
    const int hsel = n >> 6, f = n & 63;
    const float bn = bias[n];
#pragma unroll
    for (int v = 0; v < 8; ++v) {
      const int mm = mt * 16 + v + hi * 8;           // C row
      const int bb = mm >> 11, sidx = mm & (S_ - 1);
      const _Float16 hv = (_Float16)(acc[j][v] + bn);
      if (which == 2)
        vth[(((size_t)bb * H_ + hsel) * DH_ + f) * S_ + sidx] = hv;
      else if (which == 0)
        qh[(((size_t)bb * H_ + hsel) * S_ + sidx) * DH_ + f] = hv;
      else
        kh[(((size_t)bb * H_ + hsel) * S_ + sidx) * DH_ + f] = hv;
    }
  }
}

// ---------------------------------------------------------------------------
// 4) Attention: one wave per (b,h,q-tile). Two passes over the causal strip,
//    raw scores staged in the attn output region; probs + ctx in pass 2.
// ---------------------------------------------------------------------------
__global__ __launch_bounds__(256) void attn_kernel(
    const _Float16* __restrict__ qh, const _Float16* __restrict__ kh,
    const _Float16* __restrict__ vth, const int* __restrict__ kvlens,
    _Float16* __restrict__ ctxh, float* __restrict__ attn_base) {
  __shared__ _Float16 plds[8][16 * 32];   // per-wave 16x32 P staging tile

  const int wid = threadIdx.x >> 5, lane = threadIdx.x & 31;
  const int lq = lane & 15, hi = lane >> 4;
  const int job = blockIdx.x * 8 + wid;   // 0..4095
  const int bh = job >> 7;                // 0..31
  const int qt = job & 127;               // query tile
  const int b = bh >> 4, h = bh & 15;
  const int kvlen = kvlens[b];

  const _Float16* Q = qh + (size_t)bh * S_ * DH_;
  const _Float16* K = kh + (size_t)bh * S_ * DH_;
  const _Float16* V = vth + (size_t)bh * DH_ * S_;
  float* attn = attn_base + (size_t)bh * S_ * S_;

  // Q A-fragments (16x64 split into two 16x32 chunks)
  v16h aq[2];
#pragma unroll
  for (int c = 0; c < 2; ++c) {
    const _Float16* qp = Q + (size_t)(qt * 16 + lq) * DH_ + c * 32 + hi * 8;
    aq[c] = mk16(*(const v8h*)qp, *(const v8h*)(qp + 16));
  }

  float rowM[8], rowS[8];
#pragma unroll
  for (int v = 0; v < 8; ++v) { rowM[v] = -1e30f; rowS[v] = 0.f; }

  // ---- Pass 1: raw masked scores + online (max,sum) row statistics ----
  for (int kt = 0; kt <= qt; ++kt) {
    v8f sc = zero8();
#pragma unroll
    for (int c = 0; c < 2; ++c) {
      v16h bk = mk16(*(const v8h*)(K + (size_t)(kt * 16 + lq) * DH_ + c * 32 + hi * 8), *(const v8h*)(K + (size_t)(kt * 16 + lq) * DH_ + c * 32 + hi * 8 + 16));
      sc = wmma_f16(aq[c], bk, sc);
    }
    float sv[8];
#pragma unroll
    for (int v = 0; v < 8; ++v) {
      const int qg = qt * 16 + v + hi * 8;  // C row = global query index
      const int kg = kt * 16 + lq;          // C col = global key index
      float s = sc[v] * SCALE;
      s = ((kg <= qg) && (kg < kvlen)) ? s : -1e30f;
      sv[v] = s;
      attn[(size_t)qg * S_ + kg] = s;       // stage raw score
    }
#pragma unroll
    for (int v = 0; v < 8; ++v) {
      float mx = sv[v];
#pragma unroll
      for (int d = 1; d < 16; d <<= 1) mx = fmaxf(mx, __shfl_xor(mx, d, 16));
      float e = __expf(sv[v] - mx);
#pragma unroll
      for (int d = 1; d < 16; d <<= 1) e += __shfl_xor(e, d, 16);
      const float Mn = fmaxf(rowM[v], mx);
      rowS[v] = rowS[v] * __expf(rowM[v] - Mn) + e * __expf(mx - Mn);
      rowM[v] = Mn;
    }
  }

  float invS[8];
#pragma unroll
  for (int v = 0; v < 8; ++v) invS[v] = 1.0f / rowS[v];

  // ---- Pass 2: normalized probs out + ctx = P @ V via WMMA ----
  v8f cc[4] = {zero8(), zero8(), zero8(), zero8()};
  _Float16* myp = &plds[wid][0];
  const int npairs = (qt >> 1) + 1;         // pairs of 16-key tiles (K=32)
  for (int pc = 0; pc < npairs; ++pc) {
#pragma unroll
    for (int tloc = 0; tloc < 2; ++tloc) {
      const int t = pc * 2 + tloc;
      if (t <= qt) {
#pragma unroll
        for (int v = 0; v < 8; ++v) {
          const int r = v + hi * 8;
          const int qg = qt * 16 + r;
          const int kg = t * 16 + lq;
          const size_t idx = (size_t)qg * S_ + kg;
          const float p = __expf(attn[idx] - rowM[v]) * invS[v];
          attn[idx] = p;                    // overwrite score with prob
          myp[r * 32 + tloc * 16 + lq] = (_Float16)p;
        }
      } else {
#pragma unroll
        for (int v = 0; v < 8; ++v)
          myp[(v + hi * 8) * 32 + tloc * 16 + lq] = (_Float16)0.f;
      }
    }
    asm volatile("s_wait_dscnt 0" ::: "memory");  // wave-local LDS ordering
    const _Float16* pp = myp + lq * 32 + hi * 8;
    v16h ap = mk16(*(const v8h*)pp, *(const v8h*)(pp + 16));
#pragma unroll
    for (int j = 0; j < 4; ++j) {
      v16h bv = mk16(*(const v8h*)(V + (size_t)(j * 16 + lq) * S_ + pc * 32 + hi * 8), *(const v8h*)(V + (size_t)(j * 16 + lq) * S_ + pc * 32 + hi * 8 + 16));
      cc[j] = wmma_f16(ap, bv, cc[j]);
    }
  }

  // ctx -> [b, s, h*64+f] as f16 for the output projection
#pragma unroll
  for (int j = 0; j < 4; ++j) {
    const int f = j * 16 + lq;
#pragma unroll
    for (int v = 0; v < 8; ++v) {
      const int sg = qt * 16 + v + hi * 8;
      ctxh[((size_t)b * S_ + sg) * D_ + h * DH_ + f] = (_Float16)cc[j][v];
    }
  }

  // Zero strictly-future key tiles (reference attn is 0 there)
  const int c0start = (qt + 1) * 16;
#pragma unroll
  for (int v = 0; v < 8; ++v) {
    const int qg = qt * 16 + v + hi * 8;
    float4 z; z.x = z.y = z.z = z.w = 0.f;
    for (int c0 = c0start + lq * 4; c0 < S_; c0 += 64)
      *(float4*)(attn + (size_t)qg * S_ + c0) = z;
  }
}

// ---------------------------------------------------------------------------
// 5) Output projection: out = ctx @ wo^T + bo  (fp32 result)
// ---------------------------------------------------------------------------
__global__ __launch_bounds__(256) void out_gemm_kernel(
    const _Float16* __restrict__ ctxh, const _Float16* __restrict__ woh,
    const float* __restrict__ bo, float* __restrict__ out) {
  const int wid = threadIdx.x >> 5, lane = threadIdx.x & 31;
  const int lq = lane & 15, hi = lane >> 4;
  const int job = blockIdx.x * 8 + wid;
  const int mt = job >> 4, ns = job & 15;
  const int m  = mt * 16 + lq;
  const int n0 = ns * 64;

  v8f acc[4] = {zero8(), zero8(), zero8(), zero8()};
  const _Float16* arow = ctxh + (size_t)m * D_;
  for (int kt = 0; kt < D_; kt += 32) {
    const _Float16* ap = arow + kt + hi * 8;
    __builtin_prefetch(ap + 128, 0, 0);
    v16h a = mk16(*(const v8h*)ap, *(const v8h*)(ap + 16));
#pragma unroll
    for (int j = 0; j < 4; ++j) {
      const int n = n0 + j * 16 + lq;
      v16h b = mk16(*(const v8h*)(woh + (size_t)n * D_ + kt + hi * 8), *(const v8h*)(woh + (size_t)n * D_ + kt + hi * 8 + 16));
      acc[j] = wmma_f16(a, b, acc[j]);
    }
  }
#pragma unroll
  for (int j = 0; j < 4; ++j) {
    const int n = n0 + j * 16 + lq;
    const float bn = bo[n];
#pragma unroll
    for (int v = 0; v < 8; ++v) {
      const int mm = mt * 16 + v + hi * 8;
      out[(size_t)mm * D_ + n] = acc[j][v] + bn;
    }
  }
}

// ---------------------------------------------------------------------------
extern "C" void kernel_launch(void* const* d_in, const int* in_sizes, int n_in,
                              void* d_out, int out_size, void* d_ws,
                              size_t ws_size, hipStream_t stream) {
  const float* x     = (const float*)d_in[0];
  const int*   kvl   = (const int*)d_in[1];
  const float* gamma = (const float*)d_in[2];
  const float* beta  = (const float*)d_in[3];
  const float* wq = (const float*)d_in[4];
  const float* bq = (const float*)d_in[5];
  const float* wk = (const float*)d_in[6];
  const float* bk = (const float*)d_in[7];
  const float* wv = (const float*)d_in[8];
  const float* bv = (const float*)d_in[9];
  const float* wo = (const float*)d_in[10];
  const float* bo = (const float*)d_in[11];

  float* out  = (float*)d_out;
  float* attn = out + (size_t)B_ * S_ * D_;  // 512 MB attn output region

  // Workspace layout (f16 elements). Total ~48 MB.
  _Float16* ws = (_Float16*)d_ws;
  const size_t XN   = (size_t)B_ * S_ * D_;       // 4,194,304
  const size_t WSZ  = (size_t)D_ * D_;            // 1,048,576
  const size_t QKV  = (size_t)B_ * H_ * S_ * DH_; // 4,194,304
  _Float16* xnh = ws;
  _Float16* wqh = xnh + XN;
  _Float16* wkh = wqh + WSZ;
  _Float16* wvh = wkh + WSZ;
  _Float16* woh = wvh + WSZ;
  _Float16* qhp = woh + WSZ;
  _Float16* khp = qhp + QKV;
  _Float16* vth = khp + QKV;
  _Float16* ctxh = vth + QKV;

  ln_kernel<<<B_ * S_, 256, 0, stream>>>(x, gamma, beta, xnh);

  cvt_kernel<<<512, 256, 0, stream>>>(wq, wqh, (int)WSZ);
  cvt_kernel<<<512, 256, 0, stream>>>(wk, wkh, (int)WSZ);
  cvt_kernel<<<512, 256, 0, stream>>>(wv, wvh, (int)WSZ);
  cvt_kernel<<<512, 256, 0, stream>>>(wo, woh, (int)WSZ);

  // 3 matrices x 4096 wave-jobs / 8 waves per block
  qkv_gemm_kernel<<<dim3(512, 3), 256, 0, stream>>>(xnh, wqh, wkh, wvh, bq, bk,
                                                    bv, qhp, khp, vth);

  // 32 (b,h) x 128 q-tiles = 4096 wave-jobs
  attn_kernel<<<512, 256, 0, stream>>>(qhp, khp, vth, kvl, ctxh, attn);

  out_gemm_kernel<<<512, 256, 0, stream>>>(ctxh, woh, bo, out);
}
